// CFGEncoder_GGNN_82935818486047
// MI455X (gfx1250) — matrix operand order repaired and checked by reading, not compiled, not run on hardware
//
#include <hip/hip_runtime.h>
#include <hip/hip_bf16.h>

// GGNN encoder: B=16, N=768, E=3, D=64, NE=2304, 5 steps.
// All GEMMs via v_wmma_f32_16x16x32_bf16 (wave32). A converted to bf16 once
// (113 MB -> fits MI455X 192MB L2; read once per step for 5 steps).
// Dominant A@s GEMM: K-split x4 for occupancy (768 blocks), double-buffered
// LDS staging via GLOBAL_LOAD_ASYNC_TO_LDS_B128 (ASYNCcnt pipeline).

typedef __bf16 bf16;
typedef __attribute__((ext_vector_type(16))) __bf16 v16bf;
typedef __attribute__((ext_vector_type(8)))  __bf16 v8bf;
typedef __attribute__((ext_vector_type(4)))  __bf16 v4bf;
typedef __attribute__((ext_vector_type(8)))  float  v8f;

#define Bn 16
#define Nn 768
#define En 3
#define Dn 64
#define NEn 2304
#define K2NE 4608
#define KSPLIT 4
#define KCHUNKS (72 / KSPLIT)

#if __has_builtin(__builtin_amdgcn_global_load_async_to_lds_b128) && \
    __has_builtin(__builtin_amdgcn_s_wait_asynccnt)
#define HAVE_ASYNC 1
// Match the builtin's exact parameter types (from hipcc diagnostic):
//   arg0: vector_size(16) int, address_space(1) [global] pointer
//   arg1: vector_size(16) int, address_space(3) [LDS] pointer
typedef int v4i_vs __attribute__((vector_size(16)));
typedef __attribute__((address_space(1))) v4i_vs* gv4i_p;
typedef __attribute__((address_space(3))) v4i_vs* lv4i_p;
#else
#define HAVE_ASYNC 0
#endif

__device__ __forceinline__ v8f wmma_bf16(v16bf a, v16bf b, v8f c) {
  return __builtin_amdgcn_wmma_f32_16x16x32_bf16(false, a, false, b, (short)0, c,
                                                 false, false);
}

// A-operand 16x32 bf16 tile from row-major (M,K), ld in elements.
// ISA layout: lane<16: M=lane, K 0-7 (elems0-7), 16-23 (elems8-15);
//             lane>=16: M=lane-16, K 8-15 / 24-31.
__device__ __forceinline__ v16bf load_a_tile(const bf16* p, int ld) {
  const int lane = threadIdx.x & 31;
  const int m = lane & 15, sel = lane >> 4;
  const v8bf a0 = *(const v8bf*)(p + (size_t)m * ld + sel * 8);
  const v8bf a1 = *(const v8bf*)(p + (size_t)m * ld + 16 + sel * 8);
  v16bf r;
#pragma unroll
  for (int i = 0; i < 8; ++i) { r[i] = a0[i]; r[8 + i] = a1[i]; }
  return r;
}

// Same but source is f32 (convert on the fly).
__device__ __forceinline__ v16bf load_a_tile_f32(const float* p, int ld) {
  const int lane = threadIdx.x & 31;
  const int m = lane & 15, sel = lane >> 4;
  const float* q0 = p + (size_t)m * ld + sel * 8;
  const float* q1 = p + (size_t)m * ld + 16 + sel * 8;
  v16bf r;
#pragma unroll
  for (int i = 0; i < 8; ++i) { r[i] = (bf16)q0[i]; r[8 + i] = (bf16)q1[i]; }
  return r;
}

// B-operand 32x16 tile loaded from "BT" storage: row n (=output col), ld elems,
// lane<16: col=lane, K 0-7 / 16-23; lane>=16: K 8-15 / 24-31 (matches A).
__device__ __forceinline__ v16bf load_bt_tile(const bf16* p, int ld) {
  const int lane = threadIdx.x & 31;
  const int n = lane & 15, sel = lane >> 4;
  const bf16* q = p + (size_t)n * ld + sel * 8;
  const v8bf b0 = *(const v8bf*)(q);
  const v8bf b1 = *(const v8bf*)(q + 16);
  v16bf r;
#pragma unroll
  for (int i = 0; i < 8; ++i) { r[i] = b0[i]; r[8 + i] = b1[i]; }
  return r;
}

__device__ __forceinline__ float sigmoidf_(float x) {
  return 1.0f / (1.0f + __expf(-x));
}

// ---------------- prep kernels ----------------

__global__ void k_cvt_bf16(const float* __restrict__ src, bf16* __restrict__ dst,
                           long n) {
  long i = ((long)blockIdx.x * blockDim.x + threadIdx.x) * 4;
  if (i + 3 < n) {
    const float4 v = *(const float4*)(src + i);
    v4bf p;
    p[0] = (bf16)v.x; p[1] = (bf16)v.y; p[2] = (bf16)v.z; p[3] = (bf16)v.w;
    *(v4bf*)(dst + i) = p;
  } else {
    for (long k = i; k < n; ++k) dst[k] = (bf16)src[k];
  }
}

__global__ void k_init_h(const float* __restrict__ ps, float* __restrict__ hc,
                         bf16* __restrict__ hb, int n) {
  int i = blockIdx.x * blockDim.x + threadIdx.x;
  if (i < n) { float v = ps[i]; hc[i] = v; hb[i] = (bf16)v; }
}

__global__ void k_zero(float4* __restrict__ p, int n4) {
  int i = blockIdx.x * blockDim.x + threadIdx.x;
  if (i < n4) p[i] = make_float4(0.f, 0.f, 0.f, 0.f);
}

// ---------------- step kernel 1: s_in/s_out = h @ W^T + b, stored transposed
// sT[b][h-col 64][k 2304], k = e*768 + node. One wave per 16-node strip.
__global__ void k_sgemm(const bf16* __restrict__ h16,
                        const bf16* __restrict__ Win, const bf16* __restrict__ Wout,
                        const float* __restrict__ bin, const float* __restrict__ bout,
                        bf16* __restrict__ sTin, bf16* __restrict__ sTout) {
  const int wave = threadIdx.x >> 5, lane = threadIdx.x & 31;
  const int n = lane & 15, sel = lane >> 4;
  const int id = blockIdx.x * 8 + wave;          // 0..4607
  const int ntile = id % 48;
  const int e = (id / 48) % En;
  const int io = (id / 144) & 1;
  const int b = id / 288;
  const bf16* W = (io ? Wout : Win) + (size_t)e * Dn * Dn;   // (h,d) row-major = BT
  const float* bias = (io ? bout : bin) + e * Dn;
  bf16* sT = (io ? sTout : sTin) + (size_t)b * Dn * NEn;
  const bf16* hb = h16 + ((size_t)b * Nn + ntile * 16) * Dn;

  v8f acc[4];
#pragma unroll
  for (int i = 0; i < 4; ++i)
#pragma unroll
    for (int j = 0; j < 8; ++j) acc[i][j] = 0.f;

#pragma unroll
  for (int kc = 0; kc < 2; ++kc) {
    v16bf at = load_a_tile(hb + kc * 32, Dn);
#pragma unroll
    for (int nt = 0; nt < 4; ++nt)
      acc[nt] = wmma_bf16(at, load_bt_tile(W + nt * 16 * Dn + kc * 32, Dn), acc[nt]);
  }
  // transposed store: for fixed col (=lane n), rows are contiguous k values.
#pragma unroll
  for (int nt = 0; nt < 4; ++nt) {
    const int col = nt * 16 + n;
    const float bv = bias[col];
    v8bf pk;
#pragma unroll
    for (int j = 0; j < 8; ++j) pk[j] = (bf16)(acc[nt][j] + bv);
    *(v8bf*)(sT + (size_t)col * NEn + e * Nn + ntile * 16 + sel * 8) = pk;
  }
}

// ---------------- step kernel 2: a += A16 @ s, K-split x4, async LDS pipeline
__global__ void k_prop(const bf16* __restrict__ A16,
                       const bf16* __restrict__ sTin, const bf16* __restrict__ sTout,
                       float* __restrict__ a_in, float* __restrict__ a_out) {
  __shared__ __align__(16) bf16 sbuf[2][64][40];   // double-buffered 64x32 slab
  const int wave = threadIdx.x >> 5, lane = threadIdx.x & 31;
  const int n = lane & 15, sel = lane >> 4;
  int idx = blockIdx.x;
  const int split = idx & 3; idx >>= 2;
  const int ntb = idx % 6;   idx /= 6;
  const int io = idx & 1;
  const int b = idx >> 1;
  const int k0 = split * KCHUNKS;                  // chunk offset (32 k each)
  const int node0 = (ntb * 8 + wave) * 16;
  const bf16* sT = (io ? sTout : sTin) + (size_t)b * Dn * NEn;
  const bf16* Ab = A16 + ((size_t)b * Nn + node0) * K2NE + (size_t)io * NEn +
                   (size_t)k0 * 32;
  float* outp = (io ? a_out : a_in) + ((size_t)b * Nn + node0) * Dn;

  const int srow = threadIdx.x >> 2;   // 0..63
  const int spart = threadIdx.x & 3;   // 0..3
  const bf16* ssrc = sT + (size_t)srow * NEn + (size_t)k0 * 32 + spart * 8;

  v8f acc[4];
#pragma unroll
  for (int i = 0; i < 4; ++i)
#pragma unroll
    for (int j = 0; j < 8; ++j) acc[i][j] = 0.f;

  auto stage = [&](int kc, int buf) {
    const bf16* src = ssrc + kc * 32;
    bf16* dst = &sbuf[buf][srow][spart * 8];
#if HAVE_ASYNC
    __builtin_amdgcn_global_load_async_to_lds_b128(
        (gv4i_p)(uintptr_t)src, (lv4i_p)(unsigned)(uintptr_t)dst, 0, 0);
#else
    *(v8bf*)dst = *(const v8bf*)src;
#endif
  };

  stage(0, 0);
  for (int kc = 0; kc < KCHUNKS; ++kc) {
    const int p = kc & 1;
#if HAVE_ASYNC
    __builtin_amdgcn_s_wait_asynccnt(0);   // my slab DMA done
#endif
    __syncthreads();                        // everyone's DMA + prior reads done
    if (kc + 1 < KCHUNKS) stage(kc + 1, p ^ 1);
    if (kc + 8 < KCHUNKS)
      __builtin_prefetch(Ab + (size_t)(lane & 15) * K2NE + (size_t)(kc + 8) * 32,
                         0, 1);
    v16bf at = load_a_tile(Ab + kc * 32, K2NE);
#pragma unroll
    for (int nt = 0; nt < 4; ++nt)
      acc[nt] = wmma_bf16(at, load_bt_tile(&sbuf[p][nt * 16][0], 40), acc[nt]);
  }
#pragma unroll
  for (int nt = 0; nt < 4; ++nt)
#pragma unroll
    for (int j = 0; j < 8; ++j)
      __hip_atomic_fetch_add(&outp[(size_t)(j + sel * 8) * Dn + nt * 16 + n],
                             acc[nt][j], __ATOMIC_RELAXED,
                             __HIP_MEMORY_SCOPE_AGENT);
}

// ---------------- step kernel 3: GRU gates + h update -----------------------
__global__ void k_gate(const float* __restrict__ a_in, const float* __restrict__ a_out,
                       float* __restrict__ hcur, bf16* __restrict__ h16,
                       const bf16* __restrict__ Wr, const bf16* __restrict__ Wz,
                       const bf16* __restrict__ Wh,
                       const float* __restrict__ br, const float* __restrict__ bz,
                       const float* __restrict__ bh) {
  __shared__ __align__(16) bf16 rbuf[8][16][72];   // per-wave r*h tile (m,k)
  const int wave = threadIdx.x >> 5, lane = threadIdx.x & 31;
  const int n = lane & 15, sel = lane >> 4;
  const int id = blockIdx.x * 8 + wave;   // 0..767
  const int ntile = id % 48;
  const int b = id / 48;
  const size_t rowbase = ((size_t)b * Nn + ntile * 16) * Dn;
  const float* ai = a_in + rowbase;
  const float* ao = a_out + rowbase;
  float* hc = hcur + rowbase;
  bf16* hb = h16 + rowbase;

  v8f accr[4], accz[4], accu[4];
#pragma unroll
  for (int i = 0; i < 4; ++i)
#pragma unroll
    for (int j = 0; j < 8; ++j) { accr[i][j] = 0.f; accz[i][j] = 0.f; accu[i][j] = 0.f; }

  // X = [a_in | a_out | h], K = 192 -> 6 chunks of 32.
#pragma unroll
  for (int c = 0; c < 6; ++c) {
    const float* src = (c < 2) ? ai : (c < 4) ? ao : hc;
    v16bf at = load_a_tile_f32(src + (c & 1) * 32, Dn);
#pragma unroll
    for (int nt = 0; nt < 4; ++nt) {
      accr[nt] = wmma_bf16(at, load_bt_tile(Wr + nt * 16 * 192 + c * 32, 192), accr[nt]);
      accz[nt] = wmma_bf16(at, load_bt_tile(Wz + nt * 16 * 192 + c * 32, 192), accz[nt]);
      if (c < 4)
        accu[nt] = wmma_bf16(at, load_bt_tile(Wh + nt * 16 * 192 + c * 32, 192), accu[nt]);
    }
  }

  float rv[4][8], zv[4][8], hv[4][8];
#pragma unroll
  for (int nt = 0; nt < 4; ++nt) {
    const int col = nt * 16 + n;
    const float brv = br[col], bzv = bz[col];
#pragma unroll
    for (int j = 0; j < 8; ++j) {
      const int m = j + sel * 8;
      float hval = hc[(size_t)m * Dn + col];
      float r = sigmoidf_(accr[nt][j] + brv);
      float z = sigmoidf_(accz[nt][j] + bzv);
      rv[nt][j] = r; zv[nt][j] = z; hv[nt][j] = hval;
      rbuf[wave][m][col] = (bf16)(r * hval);     // C-layout -> (m,k) via LDS
    }
  }
  __syncthreads();
  // third K block of W_h uses r*h
#pragma unroll
  for (int c = 0; c < 2; ++c) {
    v16bf at = load_a_tile(&rbuf[wave][0][0] + c * 32, 72);
#pragma unroll
    for (int nt = 0; nt < 4; ++nt)
      accu[nt] = wmma_bf16(at, load_bt_tile(Wh + nt * 16 * 192 + 128 + c * 32, 192),
                           accu[nt]);
  }
#pragma unroll
  for (int nt = 0; nt < 4; ++nt) {
    const int col = nt * 16 + n;
    const float bhv = bh[col];
#pragma unroll
    for (int j = 0; j < 8; ++j) {
      const int m = j + sel * 8;
      float hh = tanhf(accu[nt][j] + bhv);
      float hn = (1.f - zv[nt][j]) * hv[nt][j] + zv[nt][j] * hh;
      hc[(size_t)m * Dn + col] = hn;
      hb[(size_t)m * Dn + col] = (bf16)hn;
    }
  }
}

// ---------------- output kernel: out = tanh([h,h0] @ Wo^T + bo) --------------
__global__ void k_outp(const bf16* __restrict__ h16, const float* __restrict__ prop,
                       const bf16* __restrict__ Wo, const float* __restrict__ bo,
                       float* __restrict__ out) {
  const int wave = threadIdx.x >> 5, lane = threadIdx.x & 31;
  const int n = lane & 15, sel = lane >> 4;
  const int id = blockIdx.x * 8 + wave;
  const int ntile = id % 48;
  const int b = id / 48;
  const size_t rowbase = ((size_t)b * Nn + ntile * 16) * Dn;

  v8f acc[4];
#pragma unroll
  for (int i = 0; i < 4; ++i)
#pragma unroll
    for (int j = 0; j < 8; ++j) acc[i][j] = 0.f;

#pragma unroll
  for (int c = 0; c < 4; ++c) {
    v16bf at = (c < 2) ? load_a_tile(h16 + rowbase + (c & 1) * 32, Dn)
                       : load_a_tile_f32(prop + rowbase + (c & 1) * 32, Dn);
#pragma unroll
    for (int nt = 0; nt < 4; ++nt)
      acc[nt] = wmma_bf16(at, load_bt_tile(Wo + nt * 16 * 128 + c * 32, 128), acc[nt]);
  }
#pragma unroll
  for (int nt = 0; nt < 4; ++nt) {
    const int col = nt * 16 + n;
    const float bv = bo[col];
#pragma unroll
    for (int j = 0; j < 8; ++j)
      out[rowbase + (size_t)(j + sel * 8) * Dn + col] = tanhf(acc[nt][j] + bv);
  }
}

// ---------------- host launcher ----------------------------------------------
extern "C" void kernel_launch(void* const* d_in, const int* in_sizes, int n_in,
                              void* d_out, int out_size, void* d_ws, size_t ws_size,
                              hipStream_t stream) {
  const float* prop = (const float*)d_in[0];
  const float* Afull = (const float*)d_in[2];
  const float* W_in = (const float*)d_in[4];
  const float* b_in = (const float*)d_in[5];
  const float* W_out = (const float*)d_in[6];
  const float* b_out = (const float*)d_in[7];
  const float* W_r = (const float*)d_in[8];
  const float* b_r = (const float*)d_in[9];
  const float* W_z = (const float*)d_in[10];
  const float* b_z = (const float*)d_in[11];
  const float* W_h = (const float*)d_in[12];
  const float* b_h = (const float*)d_in[13];
  const float* W_o = (const float*)d_in[14];
  const float* b_o = (const float*)d_in[15];
  (void)in_sizes; (void)n_in; (void)out_size; (void)ws_size;

  size_t off = 0;
  auto carve = [&](size_t bytes) -> void* {
    void* p = (char*)d_ws + off;
    off += (bytes + 255) & ~(size_t)255;
    return p;
  };
  bf16* A16   = (bf16*)carve((size_t)Bn * Nn * K2NE * 2);   // 113 MB, L2-resident
  bf16* h16   = (bf16*)carve((size_t)Bn * Nn * Dn * 2);
  float* hcur = (float*)carve((size_t)Bn * Nn * Dn * 4);
  bf16* sTin  = (bf16*)carve((size_t)Bn * Dn * NEn * 2);
  bf16* sTout = (bf16*)carve((size_t)Bn * Dn * NEn * 2);
  float* a_in = (float*)carve((size_t)Bn * Nn * Dn * 4);
  float* a_out= (float*)carve((size_t)Bn * Nn * Dn * 4);
  bf16* Win16 = (bf16*)carve((size_t)En * Dn * Dn * 2);
  bf16* Wout16= (bf16*)carve((size_t)En * Dn * Dn * 2);
  bf16* Wr16  = (bf16*)carve((size_t)Dn * 192 * 2);
  bf16* Wz16  = (bf16*)carve((size_t)Dn * 192 * 2);
  bf16* Wh16  = (bf16*)carve((size_t)Dn * 192 * 2);
  bf16* Wo16  = (bf16*)carve((size_t)Dn * 128 * 2);

  auto cvt = [&](const float* s, bf16* d, long n) {
    int grid = (int)((n / 4 + 255) / 256);
    k_cvt_bf16<<<grid, 256, 0, stream>>>(s, d, n);
  };
  cvt(Afull, A16, (long)Bn * Nn * K2NE);
  cvt(W_in, Win16, (long)En * Dn * Dn);
  cvt(W_out, Wout16, (long)En * Dn * Dn);
  cvt(W_r, Wr16, (long)Dn * 192);
  cvt(W_z, Wz16, (long)Dn * 192);
  cvt(W_h, Wh16, (long)Dn * 192);
  cvt(W_o, Wo16, (long)Dn * 128);

  k_init_h<<<(Bn * Nn * Dn + 255) / 256, 256, 0, stream>>>(prop, hcur, h16,
                                                           Bn * Nn * Dn);

  const int nzero4 = (Bn * Nn * Dn) / 4;
  for (int step = 0; step < 5; ++step) {
    k_sgemm<<<576, 256, 0, stream>>>(h16, Win16, Wout16, b_in, b_out, sTin, sTout);
    k_zero<<<(nzero4 + 255) / 256, 256, 0, stream>>>((float4*)a_in, nzero4);
    k_zero<<<(nzero4 + 255) / 256, 256, 0, stream>>>((float4*)a_out, nzero4);
    k_prop<<<Bn * 2 * 6 * KSPLIT, 256, 0, stream>>>(A16, sTin, sTout, a_in, a_out);
    k_gate<<<96, 256, 0, stream>>>(a_in, a_out, hcur, h16, Wr16, Wz16, Wh16,
                                   b_r, b_z, b_h);
  }
  k_outp<<<96, 256, 0, stream>>>(h16, prop, Wo16, b_o, (float*)d_out);
}
